// DecoderSRNN_60301340835885
// MI455X (gfx1250) — compile-verified, matrix operand order checked by reading, not run on hardware
//
#include <hip/hip_runtime.h>
#include <math.h>

// ---------------------------------------------------------------- problem dims
constexpr int B  = 4096;
constexpr int H  = 256;
constexpr int V  = 32000;
constexpr int NS = 2;
constexpr int S  = 128;
constexpr int D  = 4;
constexpr int E  = 64;
constexpr int KC = 1024;           // concat K for mid GEMM: emb|hidden|sv0|sv1
constexpr int NG = 144;            // gate GEMM cols: 128 push + 6 act + 10 pad

// output layout in d_out (floats, reference tuple order)
constexpr size_t O_OUT = 0;
constexpr size_t O_NH  = (size_t)B * V;              // 131072000
constexpr size_t O_TOP = O_NH + (size_t)B * H;       // 132120576
constexpr size_t O_STK = O_TOP + (size_t)B;          // 132124672

// ---------------------------------------------------------------- WMMA helpers
typedef _Float16 v8h  __attribute__((ext_vector_type(8)));
typedef _Float16 v16h __attribute__((ext_vector_type(16)));
typedef float    v8f  __attribute__((ext_vector_type(8)));
typedef unsigned int u32x4 __attribute__((ext_vector_type(4)));
typedef int          i32x4 __attribute__((ext_vector_type(4)));
typedef int          i32x8 __attribute__((ext_vector_type(8)));

// CDNA5 16-bit A-matrix 16x32 lane layout (ISA 7.12.2):
// lane L: row = L&15, half-group g = L>>4.
// lane's 16 halves = K[g*8 .. g*8+7] ++ K[16+g*8 .. 16+g*8+7]  (two b128 loads)
__device__ __forceinline__ v16h load_tile16(const _Float16* base, int ld,
                                            int row0, int k0, int lane) {
  int r = lane & 15, g = lane >> 4;
  const _Float16* p = base + (size_t)(row0 + r) * ld + k0 + g * 8;
  v8h lo = *(const v8h*)(p);
  v8h hi = *(const v8h*)(p + 16);
  return __builtin_shufflevector(lo, hi, 0,1,2,3,4,5,6,7,8,9,10,11,12,13,14,15);
}

__device__ __forceinline__ v8f wmma_f16(v16h a, v16h b, v8f c) {
  // D = A(16x32) * B(32x16) + C, fp32 accumulate
  return __builtin_amdgcn_wmma_f32_16x16x32_f16(false, a, false, b,
                                                (short)0, c, false, false);
}

// ---------------------------------------------------------------- prep kernels
// Wcat[h, 0:1024] = [W_in[h,:], W_hh[h,:], W_s2h[0,h,:], W_s2h[1,h,:]]  (f16)
__global__ void pack_mid_weights(const float* __restrict__ W_in,
                                 const float* __restrict__ b_in,
                                 const float* __restrict__ W_hh,
                                 const float* __restrict__ b_hh,
                                 const float* __restrict__ W_s2h,
                                 const float* __restrict__ b_s2h,
                                 _Float16* __restrict__ Wcat,
                                 float* __restrict__ bmid) {
  int h = blockIdx.x;
  for (int j = 0; j < 4; ++j) {
    int k = threadIdx.x * 4 + j;
    float v;
    if      (k < 256)  v = W_in [h * 256 + k];
    else if (k < 512)  v = W_hh [h * 256 + (k - 256)];
    else if (k < 768)  v = W_s2h[0 * H * 256 + h * 256 + (k - 512)];
    else               v = W_s2h[1 * H * 256 + h * 256 + (k - 768)];
    Wcat[(size_t)h * KC + k] = (_Float16)v;
  }
  if (threadIdx.x == 0)
    bmid[h] = b_in[h] + b_hh[h] + b_s2h[h] + b_s2h[H + h];
}

// Wg rows: 0..127 = W_push[n,e,:]; 128..133 = W_act[n,a,:]; 134..143 = 0
__global__ void pack_gate_weights(const float* __restrict__ W_push,
                                  const float* __restrict__ b_push,
                                  const float* __restrict__ W_act,
                                  const float* __restrict__ b_act,
                                  _Float16* __restrict__ Wg,
                                  float* __restrict__ bg) {
  int r = blockIdx.x, t = threadIdx.x;
  float v = 0.0f;
  if      (r < 128) v = W_push[(size_t)r * 256 + t];
  else if (r < 134) v = W_act [(size_t)(r - 128) * 256 + t];
  Wg[(size_t)r * 256 + t] = (_Float16)v;
  if (t == 0)
    bg[r] = (r < 128) ? b_push[r] : ((r < 134) ? b_act[r - 128] : 0.0f);
}

__global__ void f32_to_f16(const float* __restrict__ src,
                           _Float16* __restrict__ dst, int n) {
  int i = blockIdx.x * blockDim.x + threadIdx.x;
  if (i < n) dst[i] = (_Float16)src[i];
}

// X[b, 0:1024] = [emb_table[ids[b],:], hidden[b,:], stacks[b,0,0:4,:], stacks[b,1,0:4,:]]
__global__ void build_X(const int* __restrict__ ids,
                        const float* __restrict__ hidden,
                        const float* __restrict__ stacks,
                        const float* __restrict__ emb_table,
                        _Float16* __restrict__ X) {
  int b = blockIdx.x;
  int id = ids[b];
  _Float16* row = X + (size_t)b * KC;
  for (int j = 0; j < 4; ++j) {
    int p = threadIdx.x * 4 + j;
    float v;
    if      (p < 256) v = emb_table[(size_t)id * H + p];
    else if (p < 512) v = hidden[(size_t)b * H + (p - 256)];
    else if (p < 768) v = stacks[((size_t)b * NS + 0) * S * E + (p - 512)];
    else              v = stacks[((size_t)b * NS + 1) * S * E + (p - 768)];
    row[p] = (_Float16)v;
  }
}

// ---------------------------------------------------------------- mid GEMM + tanh
// grid: B/16 blocks, 512 threads = 16 waves, wave w -> output cols [16w,16w+16)
__global__ void __launch_bounds__(512)
mid_gemm(const _Float16* __restrict__ X, const _Float16* __restrict__ Wcat,
         const float* __restrict__ bmid, float* __restrict__ nh_out,
         _Float16* __restrict__ nh16) {
  int m0   = blockIdx.x * 16;
  int wave = threadIdx.x >> 5;
  int lane = threadIdx.x & 31;
  int n0   = wave * 16;
  v8f c = {};
  for (int k0 = 0; k0 < KC; k0 += 32) {
    v16h a = load_tile16(X,    KC, m0, k0, lane);
    v16h b = load_tile16(Wcat, KC, n0, k0, lane);
    c = wmma_f16(a, b, c);
  }
  int n = n0 + (lane & 15);
  int rb = (lane >> 4) * 8;
  float bias = bmid[n];
  for (int r = 0; r < 8; ++r) {
    int row = m0 + rb + r;
    float h = tanhf(c[r] + bias);
    nh_out[(size_t)row * H + n] = h;
    nh16 [(size_t)row * H + n] = (_Float16)h;
  }
}

// ---------------------------------------------------------------- gate GEMM
// grid: B/16 blocks, 288 threads = 9 waves, wave w -> cols [16w,16w+16) of NG=144
__global__ void __launch_bounds__(288)
gates_gemm(const _Float16* __restrict__ hid16, const _Float16* __restrict__ Wg,
           const float* __restrict__ bg, float* __restrict__ act,
           float* __restrict__ push) {
  int m0   = blockIdx.x * 16;
  int wave = threadIdx.x >> 5;
  int lane = threadIdx.x & 31;
  int n0   = wave * 16;
  v8f c = {};
  for (int k0 = 0; k0 < H; k0 += 32) {
    v16h a = load_tile16(hid16, H, m0, k0, lane);
    v16h b = load_tile16(Wg,    H, n0, k0, lane);
    c = wmma_f16(a, b, c);
  }
  int n = n0 + (lane & 15);
  int rb = (lane >> 4) * 8;
  float bias = bg[n];
  for (int r = 0; r < 8; ++r) {
    int row = m0 + rb + r;
    float v = c[r] + bias;
    if (n < 128)       push[(size_t)row * 128 + n] = tanhf(v);
    else if (n < 134)  act [(size_t)row * 6 + (n - 128)] = v;
  }
}

// ---------------------------------------------------------------- logits + log_softmax + argmax
// grid: B/16 blocks, 256 threads = 8 waves; block owns 16 rows x full V.
// Pass 1: WMMA logits, stream W_out tiles (resident in the 192MB L2), online
// (max,sumexp,argmax) in regs. Reduce lanes (shfl_xor butterfly) -> waves (LDS).
// Pass 2: normalize stored logits in place.
__global__ void __launch_bounds__(256)
logits_kernel(const _Float16* __restrict__ nh16,
              const _Float16* __restrict__ Wout16,
              const float* __restrict__ b_out,
              float* __restrict__ out, float* __restrict__ top1) {
  __shared__ float ls_m[8][16], ls_s[8][16], ls_bv[8][16], ls_lse[16];
  __shared__ int   ls_bi[8][16];

  int m0   = blockIdx.x * 16;
  int tid  = threadIdx.x;
  int wave = tid >> 5;
  int lane = tid & 31;
  int ln   = lane & 15;
  int g    = lane >> 4;

  // preload the 16x256 A tile (new_hidden) for all 8 K-steps
  v16h a[8];
  for (int kk = 0; kk < 8; ++kk)
    a[kk] = load_tile16(nh16, H, m0, kk * 32, lane);

  float mrun[8], srun[8], bval[8];
  int   bidx[8];
  for (int r = 0; r < 8; ++r) {
    mrun[r] = -3.0e38f; srun[r] = 0.0f; bval[r] = -3.0e38f; bidx[r] = 0;
  }

  for (int it = 0; it < V / 128; ++it) {
    int v0 = it * 128 + wave * 16;
    if (ln == 0 && it + 1 < V / 128)  // prefetch next W_out tile (global_prefetch)
      __builtin_prefetch(Wout16 + (size_t)(v0 + 128) * H, 0, 1);
    v8f c = {};
    for (int kk = 0; kk < 8; ++kk) {
      v16h b = load_tile16(Wout16, H, v0, kk * 32, lane);
      c = wmma_f16(a[kk], b, c);
    }
    int v = v0 + ln;
    float bo = b_out[v];
    for (int r = 0; r < 8; ++r) {
      float x = c[r] + bo;
      out[(size_t)(m0 + g * 8 + r) * V + v] = x;
      float nm = fmaxf(mrun[r], x);
      srun[r] = srun[r] * __expf(mrun[r] - nm) + __expf(x - nm);
      mrun[r] = nm;
      if (x > bval[r]) { bval[r] = x; bidx[r] = v; }
    }
  }

  // butterfly reduction across the 16 lanes sharing each row half
  for (int off = 1; off < 16; off <<= 1) {
    for (int r = 0; r < 8; ++r) {
      float m2  = __shfl_xor(mrun[r], off);
      float s2  = __shfl_xor(srun[r], off);
      float nm  = fmaxf(mrun[r], m2);
      srun[r]   = srun[r] * __expf(mrun[r] - nm) + s2 * __expf(m2 - nm);
      mrun[r]   = nm;
      float bv2 = __shfl_xor(bval[r], off);
      int   bi2 = __shfl_xor(bidx[r], off);
      if (bv2 > bval[r] || (bv2 == bval[r] && bi2 < bidx[r])) {
        bval[r] = bv2; bidx[r] = bi2;
      }
    }
  }
  if (ln == 0) {
    for (int r = 0; r < 8; ++r) {
      int R = g * 8 + r;
      ls_m [wave][R] = mrun[r];
      ls_s [wave][R] = srun[r];
      ls_bv[wave][R] = bval[r];
      ls_bi[wave][R] = bidx[r];
    }
  }
  __threadfence();   // make pass-1 logit stores visible block-wide
  __syncthreads();

  if (tid < 16) {
    float m = -3.0e38f, s = 0.0f, bv = -3.0e38f;
    int bi = 0;
    for (int w = 0; w < 8; ++w) {
      float m2 = ls_m[w][tid], s2 = ls_s[w][tid];
      float nm = fmaxf(m, m2);
      s = s * __expf(m - nm) + s2 * __expf(m2 - nm);
      m = nm;
      float bv2 = ls_bv[w][tid]; int bi2 = ls_bi[w][tid];
      if (bv2 > bv || (bv2 == bv && bi2 < bi)) { bv = bv2; bi = bi2; }
    }
    ls_lse[tid] = m + __logf(s);
    top1[m0 + tid] = (float)bi;
  }
  __syncthreads();

  for (int R = 0; R < 16; ++R) {
    float l = ls_lse[R];
    float* row = out + (size_t)(m0 + R) * V;
    for (int v = tid; v < V; v += 256) row[v] -= l;
  }
}

// ---------------------------------------------------------------- stack update
// one block per (b, ns) slab; 32KB slab staged into LDS by the Tensor Data
// Mover (D# per ISA 8.3-8.5: count=1, type=2, data_size=8B, 1-row tile of
// 4096 elements), then blended and written once.
__global__ void __launch_bounds__(256)
stack_kernel(const float* __restrict__ stacks, const float* __restrict__ act,
             const float* __restrict__ push, const float* __restrict__ empty_elem,
             float* __restrict__ out) {
  __shared__ float sl[S * E];
  int slab = blockIdx.x;
  const float* src = stacks + (size_t)slab * (S * E);

#if __has_builtin(__builtin_amdgcn_tensor_load_to_lds) && \
    __has_builtin(__builtin_amdgcn_s_wait_tensorcnt)
  if (threadIdx.x == 0) {
    unsigned lds_off = (unsigned)(size_t)(void*)sl;        // LDS byte offset
    unsigned long long ga = (unsigned long long)(size_t)src;
    u32x4 g0;
    g0.x = 1u;                                             // count=1, user mode
    g0.y = lds_off;                                        // lds_addr
    g0.z = (unsigned)(ga & 0xffffffffu);                   // global_addr[31:0]
    g0.w = (unsigned)((ga >> 32) & 0x01ffffffu) | (2u << 30); // addr[56:32]|type=2
    i32x8 g1;
    g1[0] = 3 << 16;        // data_size = 3 (8 bytes); no multicast/pad/iterate
    g1[1] = 0x10000000;     // tensor_dim0 = 4096 (8B units) low16 << 16
    g1[2] = 1 << 16;        // tensor_dim0 hi16 = 0 | tensor_dim1 = 1
    g1[3] = 0x10000000;     // tensor_dim1 hi16 = 0 | tile_dim0 = 4096
    g1[4] = 1;              // tile_dim1 = 1, tile_dim2 = 0
    g1[5] = 0x1000;         // tensor_dim0_stride = 4096
    g1[6] = 0x10000000;     // stride0 hi = 0 | tensor_dim1_stride low16 = 4096
    g1[7] = 0;
    i32x4 z4 = {0, 0, 0, 0};
#if defined(__clang_major__) && __clang_major__ >= 23
    i32x8 z8 = {0, 0, 0, 0, 0, 0, 0, 0};
    __builtin_amdgcn_tensor_load_to_lds(g0, g1, z4, z4, z8, 0);
#else
    __builtin_amdgcn_tensor_load_to_lds(g0, g1, z4, z4, 0);
#endif
    __builtin_amdgcn_s_wait_tensorcnt(0);
  }
  __syncthreads();
#else
  for (int i = threadIdx.x * 4; i < S * E; i += 1024)
    *(float4*)&sl[i] = *(const float4*)&src[i];
  __syncthreads();
#endif

  int b = slab / NS, n = slab % NS;
  float pp = act[(size_t)b * 6 + n * 3 + 0];
  float po = act[(size_t)b * 6 + n * 3 + 1];
  float pn = act[(size_t)b * 6 + n * 3 + 2];
  float* dst = out + (size_t)slab * (S * E);

  for (int i = threadIdx.x * 4; i < S * E; i += 1024) {
    int s = i >> 6;
    float4 res;
    if (s == 0) {                       // p_push * push_val
      const float4 pv = *(const float4*)&push[(size_t)b * 128 + n * 64 + (i & 63)];
      res.x = pp * pv.x; res.y = pp * pv.y; res.z = pp * pv.z; res.w = pp * pv.w;
    } else if (s == S - 1) {            // empty element
      res = *(const float4*)&empty_elem[i & 63];
    } else {                            // p_push*st[s-1] + p_pop*st[s+1] + p_noop*st[s]
      float4 cur = *(float4*)&sl[i];
      float4 dn  = *(float4*)&sl[i - 64];
      float4 up  = *(float4*)&sl[i + 64];
      res.x = pp * dn.x + po * up.x + pn * cur.x;
      res.y = pp * dn.y + po * up.y + pn * cur.y;
      res.z = pp * dn.z + po * up.z + pn * cur.z;
      res.w = pp * dn.w + po * up.w + pn * cur.w;
    }
    *(float4*)&dst[i] = res;
  }
}

// ---------------------------------------------------------------- launch
extern "C" void kernel_launch(void* const* d_in, const int* in_sizes, int n_in,
                              void* d_out, int out_size, void* d_ws, size_t ws_size,
                              hipStream_t stream) {
  (void)in_sizes; (void)n_in; (void)out_size; (void)ws_size;
  const int*   input_ids  = (const int*)  d_in[0];
  const float* hidden     = (const float*)d_in[1];
  const float* stacks     = (const float*)d_in[2];
  const float* emb_table  = (const float*)d_in[3];
  const float* W_in       = (const float*)d_in[4];
  const float* b_in       = (const float*)d_in[5];
  const float* W_hh       = (const float*)d_in[6];
  const float* b_hh       = (const float*)d_in[7];
  const float* W_s2h      = (const float*)d_in[8];
  const float* b_s2h      = (const float*)d_in[9];
  const float* W_act      = (const float*)d_in[10];
  const float* b_act      = (const float*)d_in[11];
  const float* W_push     = (const float*)d_in[12];
  const float* b_push     = (const float*)d_in[13];
  const float* W_out      = (const float*)d_in[14];
  const float* b_out      = (const float*)d_in[15];
  const float* empty_elem = (const float*)d_in[16];
  float* out = (float*)d_out;

  // workspace carve-up (256B aligned)
  char* ws = (char*)d_ws;
  size_t cur = 0;
  auto alloc = [&](size_t bytes) { void* p = ws + cur;
                                   cur = (cur + bytes + 255) & ~(size_t)255; return p; };
  _Float16* Wcat  = (_Float16*)alloc((size_t)H * KC * 2);     // 512 KB
  float*    bmid  = (float*)   alloc((size_t)H * 4);
  _Float16* Wg    = (_Float16*)alloc((size_t)NG * H * 2);
  float*    bg    = (float*)   alloc((size_t)NG * 4);
  _Float16* Wout16= (_Float16*)alloc((size_t)V * H * 2);      // 16.4 MB
  _Float16* hid16 = (_Float16*)alloc((size_t)B * H * 2);      // 2 MB
  _Float16* Xmat  = (_Float16*)alloc((size_t)B * KC * 2);     // 8 MB
  _Float16* nh16  = (_Float16*)alloc((size_t)B * H * 2);      // 2 MB
  float*    actws = (float*)   alloc((size_t)B * 6 * 4);
  float*    pushws= (float*)   alloc((size_t)B * 128 * 4);    // 2 MB

  // ---- prep
  pack_mid_weights<<<H, 256, 0, stream>>>(W_in, b_in, W_hh, b_hh, W_s2h, b_s2h,
                                          Wcat, bmid);
  pack_gate_weights<<<NG, 256, 0, stream>>>(W_push, b_push, W_act, b_act, Wg, bg);
  f32_to_f16<<<(V * H + 255) / 256, 256, 0, stream>>>(W_out, Wout16, V * H);
  f32_to_f16<<<(B * H + 255) / 256, 256, 0, stream>>>(hidden, hid16, B * H);
  build_X<<<B, 256, 0, stream>>>(input_ids, hidden, stacks, emb_table, Xmat);

  // ---- recurrent core (WMMA)
  mid_gemm<<<B / 16, 512, 0, stream>>>(Xmat, Wcat, bmid, out + O_NH, nh16);
  gates_gemm<<<B / 16, 288, 0, stream>>>(hid16, Wg, bg, actws, pushws);

  // ---- logits + fused log_softmax + argmax (WMMA, streams W_out from L2)
  logits_kernel<<<B / 16, 256, 0, stream>>>(nh16, Wout16, b_out,
                                            out + O_OUT, out + O_TOP);

  // ---- stack shift/blend (TDM async copy to LDS, single read of stacks)
  stack_kernel<<<B * NS, 256, 0, stream>>>(stacks, actws, pushws, empty_elem,
                                           out + O_STK);
}
